// EquivariantFeedForward_40596030882308
// MI455X (gfx1250) — matrix operand order repaired and checked by reading, not compiled, not run on hardware
//
#include <hip/hip_runtime.h>
#include <hip/hip_bf16.h>
#include <math.h>

// ---------------------------------------------------------------------------
// Problem constants (B=8, N=4096, D=256, MLP_DIM=1024)
// ---------------------------------------------------------------------------
#define BB      8
#define NN      4096
#define DD      256
#define MLPD    1024
#define TOKENS  (BB * NN)          // 32768
#define TILE_T  16                 // tokens per workgroup
#define NTILES  (TOKENS / TILE_T)  // 2048 workgroups
#define LDX     264                // padded bf16 row stride for 256-wide LDS tiles
#define LDHID   1032               // padded bf16 row stride for 1024-wide LDS tile

typedef __bf16 bf16;
typedef __attribute__((ext_vector_type(16))) __bf16 v16bf;
typedef __attribute__((ext_vector_type(8)))  __bf16 v8bf;
typedef __attribute__((ext_vector_type(8)))  float  v8f;

// Workspace layout (bytes): packed bf16 weights, all 1 KB aligned
#define OFF_PRE0   0u
#define OFF_PRE1   131072u
#define OFF_PROJ0  262144u
#define OFF_PROJ1  393216u
#define OFF_W10    524288u
#define OFF_W11    (524288u + 1048576u)
#define OFF_W20    (524288u + 2u * 1048576u)
#define OFF_W21    (524288u + 3u * 1048576u)

// ---------------------------------------------------------------------------
// Weight packing: fp32 row-major (K x N)  ->  bf16 WMMA-B fragments.
// Fragment layout per 32x16 (KxN) tile, per the CDNA5 ISA dense-B layout:
//   lane l: N = nt*16 + (l & 15); K = kt*32 + (l>>4)*8 + (i&7) + (i>>3)*16, i = 0..15
// Packed: dst[((nt*Kt + kt)*32 + lane)*16 + i]  => one coalesced 32B load/lane.
// ---------------------------------------------------------------------------
__global__ void pack_b_kernel(const float* __restrict__ src, bf16* __restrict__ dst,
                              int K, int N) {
  int idx = blockIdx.x * blockDim.x + threadIdx.x;
  if (idx >= K * N) return;
  int Kt   = K >> 5;
  int i    = idx & 15;
  int lane = (idx >> 4) & 31;
  int tile = idx >> 9;
  int kt   = tile % Kt;
  int nt   = tile / Kt;
  int n = nt * 16 + (lane & 15);
  int k = kt * 32 + ((lane >> 4) << 3) + (i & 7) + ((i >> 3) << 4);
  dst[idx] = (bf16)src[k * N + n];
}

// ---------------------------------------------------------------------------
// Fragment loaders
// ---------------------------------------------------------------------------
// A-fragment 16x32 bf16 from an LDS row-major tile (ISA 16-bit A layout):
//   lane<16 : M=lane,    K = k0+{0..7, 16..23}
//   lane>=16: M=lane-16, K = k0+{8..15, 24..31}
__device__ __forceinline__ v16bf load_frag_a_lds(const bf16* base, int ld,
                                                 int row0, int k0, int lane) {
  int t  = lane & 15;
  int kb = k0 + ((lane >> 4) << 3);
  const bf16* p = base + (row0 + t) * ld + kb;
  v8bf lo = *(const v8bf*)p;          // ds_load_b128
  v8bf hi = *(const v8bf*)(p + 16);   // ds_load_b128
  v16bf r;
#pragma unroll
  for (int i = 0; i < 8; ++i) { r[i] = lo[i]; r[i + 8] = hi[i]; }
  return r;
}

// B-fragment from packed global weights: one contiguous 32B chunk per lane.
__device__ __forceinline__ v16bf load_frag_b_packed(const bf16* __restrict__ base,
                                                    int numKt, int nt, int kt, int lane) {
  const bf16* p = base + ((size_t)(nt * numKt + kt) * 32 + lane) * 16;
  return *(const v16bf*)p;            // 2x global_load_b128
}

__device__ __forceinline__ v8f wmma_bf16(v16bf a, v16bf b, v8f c) {
  return __builtin_amdgcn_wmma_f32_16x16x32_bf16(false, a, false, b, (short)0, c,
                                                 false, false);
}

__device__ __forceinline__ v8f vzero8() {
  v8f z = {0.f, 0.f, 0.f, 0.f, 0.f, 0.f, 0.f, 0.f};
  return z;
}

// ---------------------------------------------------------------------------
// Fused kernel: one workgroup = 16 tokens, 8 wave32 (256 threads).
//   stage X : load x tile -> bf16 LDS rows (m*16+t) x 256
//   stage 1 : pm = x @ pre (WMMA), inv = sqrt(sum_m pm^2) -> LDS bf16
//   stage 2 : hid = silu([h|inv] @ w1 + b1) -> LDS bf16
//   stage 3 : out = hid @ w2 + b2 ; m1 -> h accumulator, m2 -> LDS f32
//   stage 4 : p = x @ proj (WMMA); x_out = x + p * m2
// ---------------------------------------------------------------------------
__global__ __launch_bounds__(256)
void eff_fused_kernel(const float* __restrict__ h,
                      const float* __restrict__ x1, const float* __restrict__ x2,
                      float* __restrict__ outH,
                      float* __restrict__ outX1, float* __restrict__ outX2,
                      const bf16* __restrict__ pre0P,  const bf16* __restrict__ pre1P,
                      const bf16* __restrict__ proj0P, const bf16* __restrict__ proj1P,
                      const bf16* __restrict__ w10P,   const bf16* __restrict__ w11P,
                      const bf16* __restrict__ w20P,   const bf16* __restrict__ w21P,
                      const float* __restrict__ b1_0,  const float* __restrict__ b1_1,
                      const float* __restrict__ b2_0,  const float* __restrict__ b2_1) {
  __shared__ __align__(16) bf16  ldsX[80 * LDX];          // 42240 B (x tile, m-major)
  __shared__ __align__(16) bf16  ldsH[TILE_T * LDX];      //  8448 B (h tile, bf16)
  __shared__ __align__(16) bf16  ldsINV[TILE_T * LDX];    //  8448 B (inv tile, bf16)
  __shared__ __align__(16) bf16  ldsHID[TILE_T * LDHID];  // 33024 B (mlp hidden)
  __shared__ __align__(16) float ldsHacc[TILE_T * DD];    // 16384 B (h + m1a + m1b)
  __shared__ __align__(16) float ldsM2[TILE_T * DD];      // 16384 B (m2 gate)

  const int  tid    = threadIdx.x;
  const int  lane   = tid & 31;
  const int  wid    = tid >> 5;
  const long token0 = (long)blockIdx.x * TILE_T;

  // ---- load h tile: fp32 -> bf16 (mlp input) + fp32 accumulator ----
  {
    const float4* hsrc = (const float4*)(h + token0 * DD);
#pragma unroll
    for (int it = 0; it < 4; ++it) {
      int    i4 = tid + it * 256;   // 1024 float4 per tile
      float4 v  = hsrc[i4];
      int flat = i4 << 2;
      int t = flat >> 8, d = flat & 255;
      ldsH[t * LDX + d + 0] = (bf16)v.x;
      ldsH[t * LDX + d + 1] = (bf16)v.y;
      ldsH[t * LDX + d + 2] = (bf16)v.z;
      ldsH[t * LDX + d + 3] = (bf16)v.w;
      ldsHacc[t * DD + d + 0] = v.x;
      ldsHacc[t * DD + d + 1] = v.y;
      ldsHacc[t * DD + d + 2] = v.z;
      ldsHacc[t * DD + d + 3] = v.w;
    }
  }

  const float* xinA[2]  = {x1, x2};
  float*       xoutA[2] = {outX1, outX2};
  const bf16*  preA[2]  = {pre0P, pre1P};
  const bf16*  projA[2] = {proj0P, proj1P};
  const bf16*  w1A[2]   = {w10P, w11P};
  const bf16*  w2A[2]   = {w20P, w21P};
  const float* b1A[2]   = {b1_0, b1_1};
  const float* b2A[2]   = {b2_0, b2_1};
  const int    MA[2]    = {3, 5};

  for (int deg = 0; deg < 2; ++deg) {
    const int    M    = MA[deg];
    const float* xin  = xinA[deg];
    float*       xout = xoutA[deg];

    // ---- stage X: coalesced fp32 load of x tile -> bf16 LDS (m*16+t, d) ----
    {
      const float4* xs = (const float4*)(xin + token0 * DD * M);
      const int n4 = (TILE_T * DD * M) >> 2;  // 3072 (M=3) or 5120 (M=5)
      for (int i4 = tid; i4 < n4; i4 += 256) {
        float4 v = xs[i4];
        float  c[4] = {v.x, v.y, v.z, v.w};
        int flat = i4 << 2;
#pragma unroll
        for (int j = 0; j < 4; ++j) {
          int f  = flat + j;
          int t  = f / (DD * M);
          int rr = f - t * (DD * M);
          int d  = rr / M;
          int m  = rr - d * M;
          ldsX[(m * TILE_T + t) * LDX + d] = (bf16)c[j];
        }
      }
    }
    __syncthreads();

    // ---- stage 1: pm GEMM + inv ----
    {
      const bf16* prep = preA[deg];
#pragma unroll
      for (int cB = 0; cB < 2; ++cB) {
        int et = wid * 2 + cB;  // e-column tile 0..15
        v8f ssq = vzero8();
        for (int m = 0; m < M; ++m) {
          v8f acc = vzero8();
#pragma unroll
          for (int kt = 0; kt < 8; ++kt) {
            v16bf a = load_frag_a_lds(ldsX, LDX, m * TILE_T, kt * 32, lane);
            v16bf b = load_frag_b_packed(prep, 8, et, kt, lane);
            acc = wmma_bf16(a, b, acc);
          }
          ssq += acc * acc;
        }
        int nl = lane & 15, tb = (lane >> 4) << 3;
#pragma unroll
        for (int g = 0; g < 8; ++g)
          ldsINV[(tb + g) * LDX + et * 16 + nl] = (bf16)sqrtf(ssq[g]);
      }
    }
    __syncthreads();

    // ---- stage 2: hid = silu([h|inv] @ w1 + b1) ----
    {
      const bf16*  w1p = w1A[deg];
      const float* b1p = b1A[deg];
      for (int j = 0; j < 8; ++j) {
        int nt = wid * 8 + j;  // 0..63
        __builtin_prefetch(w1p + ((size_t)((nt + 1) & 63) * 16 * 32) * 16, 0, 0);
        v8f acc = vzero8();
#pragma unroll
        for (int kt = 0; kt < 16; ++kt) {
          v16bf a = (kt < 8)
                        ? load_frag_a_lds(ldsH, LDX, 0, kt * 32, lane)
                        : load_frag_a_lds(ldsINV, LDX, 0, (kt - 8) * 32, lane);
          v16bf b = load_frag_b_packed(w1p, 16, nt, kt, lane);
          acc = wmma_bf16(a, b, acc);
        }
        int nl = lane & 15, tb = (lane >> 4) << 3;
        int n  = nt * 16 + nl;
        float bias = b1p[n];
#pragma unroll
        for (int g = 0; g < 8; ++g) {
          float v = acc[g] + bias;
          float s = v / (1.f + __expf(-v));  // silu
          ldsHID[(tb + g) * LDHID + n] = (bf16)s;
        }
      }
    }
    __syncthreads();

    // ---- stage 3: out = hid @ w2 + b2 ; split m1 / m2 ----
    {
      const bf16*  w2p = w2A[deg];
      const float* b2p = b2A[deg];
      for (int j = 0; j < 4; ++j) {
        int nt = wid * 4 + j;  // 0..31
        v8f acc = vzero8();
#pragma unroll
        for (int kt = 0; kt < 32; ++kt) {
          v16bf a = load_frag_a_lds(ldsHID, LDHID, 0, kt * 32, lane);
          v16bf b = load_frag_b_packed(w2p, 32, nt, kt, lane);
          acc = wmma_bf16(a, b, acc);
        }
        int nl = lane & 15, tb = (lane >> 4) << 3;
        int n  = nt * 16 + nl;
        float bias = b2p[n];
        if (n < DD) {  // m1 -> residual accumulator (wave-uniform branch)
#pragma unroll
          for (int g = 0; g < 8; ++g) ldsHacc[(tb + g) * DD + n] += acc[g] + bias;
        } else {       // m2 gate
#pragma unroll
          for (int g = 0; g < 8; ++g) ldsM2[(tb + g) * DD + (n - DD)] = acc[g] + bias;
        }
      }
    }
    __syncthreads();

    // ---- stage 4: p GEMM; x_out = x + p * m2 ----
    {
      const bf16* projp = projA[deg];
      for (int idx = wid; idx < M * 16; idx += 8) {  // M*16 tiles, divisible by 8
        int m  = idx % M;
        int et = idx / M;
        v8f acc = vzero8();
#pragma unroll
        for (int kt = 0; kt < 8; ++kt) {
          v16bf a = load_frag_a_lds(ldsX, LDX, m * TILE_T, kt * 32, lane);
          v16bf b = load_frag_b_packed(projp, 8, et, kt, lane);
          acc = wmma_bf16(a, b, acc);
        }
        int nl = lane & 15, tb = (lane >> 4) << 3;
        int e  = et * 16 + nl;
#pragma unroll
        for (int g = 0; g < 8; ++g) {
          int    t   = tb + g;
          float  mod = acc[g] * ldsM2[t * DD + e];
          size_t gi  = ((size_t)(token0 + t) * DD + e) * M + m;
          xout[gi] = xin[gi] + mod;  // x tile is L2-hot
        }
      }
    }
    __syncthreads();
  }

  // ---- write h_new = h + m1_a + m1_b ----
  for (int i = tid; i < TILE_T * DD; i += 256) {
    outH[token0 * DD + i] = ldsHacc[(i >> 8) * DD + (i & 255)];
  }
}

// ---------------------------------------------------------------------------
extern "C" void kernel_launch(void* const* d_in, const int* in_sizes, int n_in,
                              void* d_out, int out_size, void* d_ws, size_t ws_size,
                              hipStream_t stream) {
  (void)in_sizes; (void)n_in; (void)out_size; (void)ws_size;

  const float* h     = (const float*)d_in[0];
  const float* x1    = (const float*)d_in[1];
  const float* x2    = (const float*)d_in[2];
  const float* proj1 = (const float*)d_in[3];
  const float* proj2 = (const float*)d_in[4];
  const float* pre1  = (const float*)d_in[5];
  const float* pre2  = (const float*)d_in[6];
  const float* w1_0  = (const float*)d_in[7];
  const float* b1_0  = (const float*)d_in[8];
  const float* w2_0  = (const float*)d_in[9];
  const float* b2_0  = (const float*)d_in[10];
  const float* w1_1  = (const float*)d_in[11];
  const float* b1_1  = (const float*)d_in[12];
  const float* w2_1  = (const float*)d_in[13];
  const float* b2_1  = (const float*)d_in[14];

  char* ws = (char*)d_ws;
  bf16* preP0  = (bf16*)(ws + OFF_PRE0);
  bf16* preP1  = (bf16*)(ws + OFF_PRE1);
  bf16* projP0 = (bf16*)(ws + OFF_PROJ0);
  bf16* projP1 = (bf16*)(ws + OFF_PROJ1);
  bf16* w1P0   = (bf16*)(ws + OFF_W10);
  bf16* w1P1   = (bf16*)(ws + OFF_W11);
  bf16* w2P0   = (bf16*)(ws + OFF_W20);
  bf16* w2P1   = (bf16*)(ws + OFF_W21);

  // Pack weights -> bf16 WMMA-B fragment layout (tiny; runs once per launch).
  pack_b_kernel<<<256, 256, 0, stream>>>(pre1, preP0, DD, DD);
  pack_b_kernel<<<256, 256, 0, stream>>>(pre2, preP1, DD, DD);
  pack_b_kernel<<<256, 256, 0, stream>>>(proj1, projP0, DD, DD);
  pack_b_kernel<<<256, 256, 0, stream>>>(proj2, projP1, DD, DD);
  pack_b_kernel<<<2048, 256, 0, stream>>>(w1_0, w1P0, 2 * DD, MLPD);
  pack_b_kernel<<<2048, 256, 0, stream>>>(w1_1, w1P1, 2 * DD, MLPD);
  pack_b_kernel<<<2048, 256, 0, stream>>>(w2_0, w2P0, MLPD, 2 * DD);
  pack_b_kernel<<<2048, 256, 0, stream>>>(w2_1, w2P1, MLPD, 2 * DD);

  float* outH  = (float*)d_out;
  float* outX1 = outH + (size_t)TOKENS * DD;          // after h_new
  float* outX2 = outX1 + (size_t)TOKENS * DD * 3;     // after x1_out

  eff_fused_kernel<<<NTILES, 256, 0, stream>>>(
      h, x1, x2, outH, outX1, outX2,
      preP0, preP1, projP0, projP1, w1P0, w1P1, w2P0, w2P1,
      b1_0, b1_1, b2_0, b2_1);
}
